// VectorQuantizer_816043786769
// MI455X (gfx1250) — matrix operand order repaired and checked by reading, not compiled, not run on hardware
//
#include <hip/hip_runtime.h>

typedef __attribute__((ext_vector_type(16))) __bf16 v16bf;
typedef __attribute__((ext_vector_type(8)))  float  v8f;

#define EMB_DIM          64
#define N_CODES          512
#define N_ROWS           (32 * 64 * 64)      // 131072 flattened pixels
#define PLANE            4096                // H*W
#define IMG_STRIDE       (EMB_DIM * PLANE)   // 262144 floats per image
#define N_CHUNKS         (N_ROWS / 16)       // 8192 row-chunks of 16
#define BLOCK_THREADS    256
#define WAVES_PER_BLOCK  8
#define CHUNKS_PER_WAVE  4                   // processed as 2 pairs
#define MAIN_BLOCKS      (N_CHUNKS / (WAVES_PER_BLOCK * CHUNKS_PER_WAVE)) // 256

union Frag { uint4 q[2]; v16bf v; };

// ---- bf16 split helpers (round-to-nearest-even) ----
__device__ __forceinline__ unsigned short f2b(float f) {
    unsigned u = __float_as_uint(f);
    unsigned r = u + 0x7FFFu + ((u >> 16) & 1u);
    return (unsigned short)(r >> 16);
}
__device__ __forceinline__ float b2f(unsigned short h) {
    return __uint_as_float(((unsigned)h) << 16);
}

// ============================================================================
// Prep kernel 1: codebook -> bf16 hi/lo (row-major [code][dim]) + bias = 0.5*||e||^2.
// Also zeroes the loss slot (d_out is poisoned by the harness).
// ============================================================================
__global__ __launch_bounds__(256) void vq_prep_w_kernel(
    const float* __restrict__ w,
    unsigned short* __restrict__ whi, unsigned short* __restrict__ wlo,
    float* __restrict__ wbias, float* __restrict__ loss)
{
    if (blockIdx.x == 0 && threadIdx.x == 0) *loss = 0.f;
    const int c = blockIdx.x * 256 + threadIdx.x;
    if (c >= N_CODES) return;
    float s = 0.f;
#pragma unroll
    for (int g = 0; g < EMB_DIM / 8; ++g) {
        unsigned hp[4], lp[4];
#pragma unroll
        for (int j = 0; j < 4; ++j) {
            unsigned short hs[2], ls[2];
#pragma unroll
            for (int e = 0; e < 2; ++e) {
                float x = w[c * EMB_DIM + g * 8 + j * 2 + e];
                s += x * x;
                unsigned short h = f2b(x);
                hs[e] = h;
                ls[e] = f2b(x - b2f(h));
            }
            hp[j] = (unsigned)hs[0] | ((unsigned)hs[1] << 16);
            lp[j] = (unsigned)ls[0] | ((unsigned)ls[1] << 16);
        }
        ((uint4*)(whi + c * EMB_DIM))[g] = make_uint4(hp[0], hp[1], hp[2], hp[3]);
        ((uint4*)(wlo + c * EMB_DIM))[g] = make_uint4(lp[0], lp[1], lp[2], lp[3]);
    }
    wbias[c] = 0.5f * s;
}

// ============================================================================
// Prep kernel 2: NCHW f32 input -> row-major [n][d] bf16 hi/lo + per-row ||x||^2.
// One thread per row n: reads are half-wave coalesced (consecutive n = consecutive
// spatial p within a channel plane), writes are contiguous 16B per thread.
// ============================================================================
__global__ __launch_bounds__(256) void vq_prep_x_kernel(
    const float* __restrict__ in,
    unsigned short* __restrict__ xhi, unsigned short* __restrict__ xlo,
    float* __restrict__ xnorm)
{
    const int n = blockIdx.x * 256 + threadIdx.x;
    const int b = n >> 12;
    const int p = n & (PLANE - 1);
    const float* src = in + (size_t)b * IMG_STRIDE + p;
    float nrm = 0.f;
#pragma unroll
    for (int g = 0; g < EMB_DIM / 8; ++g) {
        unsigned hp[4], lq[4];
#pragma unroll
        for (int j = 0; j < 4; ++j) {
            unsigned short hs[2], ls[2];
#pragma unroll
            for (int e = 0; e < 2; ++e) {
                float x = src[(size_t)(g * 8 + j * 2 + e) * PLANE];
                nrm += x * x;
                unsigned short h = f2b(x);
                hs[e] = h;
                ls[e] = f2b(x - b2f(h));
            }
            hp[j] = (unsigned)hs[0] | ((unsigned)hs[1] << 16);
            lq[j] = (unsigned)ls[0] | ((unsigned)ls[1] << 16);
        }
        ((uint4*)(xhi + (size_t)n * EMB_DIM))[g] = make_uint4(hp[0], hp[1], hp[2], hp[3]);
        ((uint4*)(xlo + (size_t)n * EMB_DIM))[g] = make_uint4(lq[0], lq[1], lq[2], lq[3]);
    }
    xnorm[n] = nrm;
}

// ============================================================================
// Main kernel: each wave owns TWO 16-row chunks at a time (B-fragment reuse:
// 12 WMMA per 8 ds_load_b128). score = x.e - 0.5||e||^2 via bf16x3 WMMA.
// Per-lane running max with the 9-bit code index packed into the low mantissa
// bits of the float score; single cross-lane butterfly per chunk at the end.
// ============================================================================
__global__ __launch_bounds__(BLOCK_THREADS) void vq_main_kernel(
    const unsigned short* __restrict__ xhi,
    const unsigned short* __restrict__ xlo,
    const float* __restrict__ xnorm,
    const unsigned short* __restrict__ whi_g,
    const unsigned short* __restrict__ wlo_g,
    const float* __restrict__ wbias_g,
    const float* __restrict__ weight,
    float* __restrict__ out,
    float* __restrict__ loss)
{
    __shared__ unsigned short s_whi[N_CODES * EMB_DIM]; // 64 KB
    __shared__ unsigned short s_wlo[N_CODES * EMB_DIM]; // 64 KB
    __shared__ float          s_bias[N_CODES];          // 2 KB
    __shared__ int            s_idx[WAVES_PER_BLOCK][16];

    { // stage codebook into LDS (once per block, amortized over 32 chunks)
        const uint4* gh = (const uint4*)whi_g;
        const uint4* gl = (const uint4*)wlo_g;
        uint4* lh = (uint4*)s_whi;
        uint4* ll = (uint4*)s_wlo;
        for (int i = threadIdx.x; i < N_CODES * EMB_DIM / 8; i += BLOCK_THREADS) {
            lh[i] = gh[i];
            ll[i] = gl[i];
        }
        for (int i = threadIdx.x; i < N_CODES; i += BLOCK_THREADS) s_bias[i] = wbias_g[i];
    }
    __syncthreads();

    const int wave = threadIdx.x >> 5;
    const int lane = threadIdx.x & 31;
    const int col  = lane & 15;   // matrix column / row-within-16 for this lane
    const int half = lane >> 4;   // which K-half this lane holds

    for (int cc = 0; cc < CHUNKS_PER_WAVE / 2; ++cc) {
        const int chunk0 = (blockIdx.x * WAVES_PER_BLOCK + wave) * CHUNKS_PER_WAVE + cc * 2;

        // ---- A fragments for both chunks (resident for the whole code sweep) ----
        // 16-bit A 16x32 layout: lanes 0-15 hold M=lane, K={kb..kb+7, kb+16..kb+23},
        // kb = 8*(lane>>4). Two b128 loads per fragment.
        v16bf ahi[2][2], alo[2][2];
        {
            const int kb = half * 8;
#pragma unroll
            for (int u = 0; u < 2; ++u) {
                const int arow = (chunk0 + u) * 16 + col;
#pragma unroll
                for (int s = 0; s < 2; ++s) {
                    Frag fh, fl;
                    const unsigned short* ph = xhi + (size_t)arow * EMB_DIM + s * 32 + kb;
                    const unsigned short* pl = xlo + (size_t)arow * EMB_DIM + s * 32 + kb;
                    fh.q[0] = *(const uint4*)(ph);
                    fh.q[1] = *(const uint4*)(ph + 16);
                    fl.q[0] = *(const uint4*)(pl);
                    fl.q[1] = *(const uint4*)(pl + 16);
                    ahi[u][s] = fh.v;
                    alo[u][s] = fl.v;
                }
            }
        }

        float bestF[2][8];
#pragma unroll
        for (int u = 0; u < 2; ++u)
#pragma unroll
            for (int r = 0; r < 8; ++r) bestF[u][r] = -INFINITY;

        for (int t = 0; t < N_CODES / 16; ++t) {
            v8f acc0 = {0.f, 0.f, 0.f, 0.f, 0.f, 0.f, 0.f, 0.f};
            v8f acc1 = {0.f, 0.f, 0.f, 0.f, 0.f, 0.f, 0.f, 0.f};
#pragma unroll
            for (int s = 0; s < 2; ++s) {
                // B 32x16 layout: lane holds column l%16, K = {kb..kb+7, kb+16..kb+23}, kb = 8*(l>>4).
                Frag bh, bl;
                const int brow = t * 16 + col;
                const int kofs = s * 32 + half * 8;
                const unsigned short* ph = s_whi + brow * EMB_DIM + kofs;
                const unsigned short* pl = s_wlo + brow * EMB_DIM + kofs;
                bh.q[0] = *(const uint4*)(ph);
                bh.q[1] = *(const uint4*)(ph + 16);
                bl.q[0] = *(const uint4*)(pl);
                bl.q[1] = *(const uint4*)(pl + 16);
                // x.w ~= hi*hi + hi*lo + lo*hi  (fp32-grade via 3x bf16 WMMA)
                acc0 = __builtin_amdgcn_wmma_f32_16x16x32_bf16(false, ahi[0][s], false, bh.v, (short)0, acc0, false, false);
                acc1 = __builtin_amdgcn_wmma_f32_16x16x32_bf16(false, ahi[1][s], false, bh.v, (short)0, acc1, false, false);
                acc0 = __builtin_amdgcn_wmma_f32_16x16x32_bf16(false, ahi[0][s], false, bl.v, (short)0, acc0, false, false);
                acc1 = __builtin_amdgcn_wmma_f32_16x16x32_bf16(false, ahi[1][s], false, bl.v, (short)0, acc1, false, false);
                acc0 = __builtin_amdgcn_wmma_f32_16x16x32_bf16(false, alo[0][s], false, bh.v, (short)0, acc0, false, false);
                acc1 = __builtin_amdgcn_wmma_f32_16x16x32_bf16(false, alo[1][s], false, bh.v, (short)0, acc1, false, false);
            }
            const float bias = s_bias[t * 16 + col];
            const unsigned code = (unsigned)(t * 16 + col);
            // pack code index into low 9 mantissa bits, keep per-lane running max
#pragma unroll
            for (int r = 0; r < 8; ++r) {
                float sc0 = acc0[r] - bias;
                float sc1 = acc1[r] - bias;
                float k0 = __uint_as_float((__float_as_uint(sc0) & 0xFFFFFE00u) | code);
                float k1 = __uint_as_float((__float_as_uint(sc1) & 0xFFFFFE00u) | code);
                bestF[0][r] = fmaxf(bestF[0][r], k0);
                bestF[1][r] = fmaxf(bestF[1][r], k1);
            }
        }

        // ---- one cross-lane butterfly per chunk (within 16-lane halves) ----
#pragma unroll
        for (int u = 0; u < 2; ++u)
#pragma unroll
            for (int r = 0; r < 8; ++r) {
                float k = bestF[u][r];
#pragma unroll
                for (int m = 8; m >= 1; m >>= 1)
                    k = fmaxf(k, __shfl_xor(k, m, 32));
                bestF[u][r] = k;
            }

        float lp = 0.f;
#pragma unroll
        for (int u = 0; u < 2; ++u) {
            const int n0 = (chunk0 + u) * 16;
            // publish per-row argmax (row M = r + 8*half)
            if (col == 0) {
#pragma unroll
                for (int r = 0; r < 8; ++r)
                    s_idx[wave][r + 8 * half] = (int)(__float_as_uint(bestF[u][r]) & 511u);
            }
            // loss: sum(||x||^2 - 2*best_score)
            if (col == 0) {
#pragma unroll
                for (int r = 0; r < 8; ++r) {
                    float bv = __uint_as_float(__float_as_uint(bestF[u][r]) & 0xFFFFFE00u);
                    lp += xnorm[n0 + r + 8 * half] - 2.0f * bv;
                }
            }
            // scatter quantized rows back to NCHW (half-wave coalesced 64B stores)
            const int b = n0 >> 12;
            const int p = n0 & (PLANE - 1);
            const int code = s_idx[wave][col];      // same-wave DS ordering
            const float* wrow = weight + code * EMB_DIM;
            float* obase = out + (size_t)b * IMG_STRIDE + p + col;
#pragma unroll 4
            for (int j = 0; j < 32; ++j) {
                const int dd = half + 2 * j;
                obase[(size_t)dd * PLANE] = wrow[dd];
            }
        }
        lp += __shfl_xor(lp, 16, 32);
        if (lane == 0) atomicAdd(loss, lp * (1.0f / (float)(N_ROWS * EMB_DIM)));
    }
}

extern "C" void kernel_launch(void* const* d_in, const int* in_sizes, int n_in,
                              void* d_out, int out_size, void* d_ws, size_t ws_size,
                              hipStream_t stream) {
    (void)in_sizes; (void)n_in; (void)ws_size;
    const float* inputs = (const float*)d_in[0];
    const float* weight = (const float*)d_in[1];
    float* out  = (float*)d_out;
    float* loss = out + (out_size - 1);   // quantized NCHW then scalar loss

    char* ws = (char*)d_ws;
    const size_t XHI  = 0;
    const size_t XLO  = XHI  + (size_t)N_ROWS * EMB_DIM * 2;   // +16 MB
    const size_t XNRM = XLO  + (size_t)N_ROWS * EMB_DIM * 2;   // +16 MB
    const size_t WHI  = XNRM + (size_t)N_ROWS * 4;             // +512 KB
    const size_t WLO  = WHI  + (size_t)N_CODES * EMB_DIM * 2;  // +64 KB
    const size_t WBIA = WLO  + (size_t)N_CODES * EMB_DIM * 2;  // +64 KB

    unsigned short* xhi   = (unsigned short*)(ws + XHI);
    unsigned short* xlo   = (unsigned short*)(ws + XLO);
    float*          xnorm = (float*)(ws + XNRM);
    unsigned short* whi   = (unsigned short*)(ws + WHI);
    unsigned short* wlo   = (unsigned short*)(ws + WLO);
    float*          wbias = (float*)(ws + WBIA);

    vq_prep_w_kernel<<<(N_CODES + 255) / 256, 256, 0, stream>>>(weight, whi, wlo, wbias, loss);
    vq_prep_x_kernel<<<N_ROWS / 256, 256, 0, stream>>>(inputs, xhi, xlo, xnorm);
    vq_main_kernel<<<MAIN_BLOCKS, BLOCK_THREADS, 0, stream>>>(xhi, xlo, xnorm, whi, wlo, wbias, weight, out, loss);
}
